// TestGCN_73504070303824
// MI455X (gfx1250) — matrix operand order repaired and checked by reading, not compiled, not run on hardware
//
#include <hip/hip_runtime.h>

// ---------------------------------------------------------------------------
// GCN (2x GCNConv + linear + softmax) for MI455X / gfx1250, wave32, WMMA.
// ---------------------------------------------------------------------------

typedef __attribute__((ext_vector_type(16))) __bf16          v16bf;
typedef __attribute__((ext_vector_type(16))) unsigned short  v16u16;
typedef __attribute__((ext_vector_type(8)))  float           v8f;

#define D 128   // D_IN == D_HID
#define NCLS 8

__device__ __forceinline__ unsigned int f2bf_bits(float f) {
    unsigned int u = __builtin_bit_cast(unsigned int, f);
    u += 0x7FFFu + ((u >> 16) & 1u);              // round-to-nearest-even
    return u >> 16;
}
__device__ __forceinline__ unsigned short f2bf(float f) {
    return (unsigned short)f2bf_bits(f);
}
__device__ __forceinline__ unsigned int pack_bf2(float lo, float hi) {
    return f2bf_bits(lo) | (f2bf_bits(hi) << 16);
}

union frag16 {
    v16u16 v;
    uint4  q[2];
};

// ---------------------------------------------------------------------------
// fill kernels
// ---------------------------------------------------------------------------
__global__ void gcn_fill_f32(float* __restrict__ p, float v, long long n) {
    long long t = (long long)blockIdx.x * blockDim.x + threadIdx.x;
    if (t < n) p[t] = v;
}

__global__ void gcn_zero_f32x4(float4* __restrict__ p, long long n4) {
    long long t = (long long)blockIdx.x * blockDim.x + threadIdx.x;
    if (t < n4) p[t] = make_float4(0.f, 0.f, 0.f, 0.f);
}

// ---------------------------------------------------------------------------
// degree (in-degree over dst; deg pre-initialized to 1.0 for self loop)
// ---------------------------------------------------------------------------
__global__ void gcn_degree(const int* __restrict__ dst, float* __restrict__ deg,
                           long long E) {
    long long t = (long long)blockIdx.x * blockDim.x + threadIdx.x;
    if (t < E) atomicAdd(&deg[dst[t]], 1.0f);
}

__global__ void gcn_rsqrt_inplace(float* __restrict__ d, long long n) {
    long long t = (long long)blockIdx.x * blockDim.x + threadIdx.x;
    if (t < n) {
        float v = d[t];
        d[t] = (v > 0.f) ? rsqrtf(v) : 0.f;
    }
}

// ---------------------------------------------------------------------------
// C[M x 128] = A[M x 128] @ W[128 x 128]     (f32 in, bf16 WMMA, f32 out)
//
// block = 256 threads (8 waves); block computes 32 rows x 128 cols.
// Wave w: row-tile mi = w&1, col-tiles j = (w>>1) and (w>>1)+4.
// K loop = 4 steps of 32 (v_wmma_f32_16x16x32_bf16), A fragment shared
// across both col-tiles.
//
// W is staged in LDS PRE-SWIZZLED into B-fragment order so each lane's
// 16-element fragment is 32 contiguous bytes -> 2x ds_load_b128.  A panel is
// row-major: fragment halves are K-contiguous -> 2x ds_load_b128.
// Bounds checks are hoisted into one uniform scalar branch (rowBase is
// block-uniform), so the hot path has no exec-mask churn.
//
// Fragment k order used by this code (16-bit A 16x32, B 32x16; same order for A and B):
//   A: lane<16 row=lane,  K=kk+e;           lane>=16: K=kk+16+e
//   B: lane<16 col=lane,  K=kk+e;           lane>=16: K=kk+16+e
// ---------------------------------------------------------------------------
__global__ __launch_bounds__(256)
void gcn_gemm_wmma(const float* __restrict__ A, const float* __restrict__ W,
                   float* __restrict__ C, int M) {
    __shared__ unsigned short sB[D * D];   // bf16 bits, fragment-order, 32 KB
    __shared__ unsigned short sA[32 * D];  // bf16 bits, row-major,       8 KB

    const int tid      = threadIdx.x;
    const int rowBase  = blockIdx.x * 32;
    const bool fullTile = (rowBase + 32 <= M);   // uniform across block

    // stage W -> fragment order: idx = ((j*4 + kk32)*32 + lane)*16 + e
    for (int i = tid; i < D * D; i += 256) {
        int k = i >> 7, n = i & (D - 1);
        float v = W[i];                       // coalesced global read
        int j = n >> 4, lmod = n & 15;
        int kk32 = k >> 5, kr = k & 31;
        int laneHi = kr >> 4, e = kr & 15;
        sB[(((j << 2) + kk32) * 32 + laneHi * 16 + lmod) * 16 + e] = f2bf(v);
    }
    // stage A panel row-major (32 x 128): float4 loads, packed bf16x2 stores
    if (fullTile) {
        const float4* A4 = (const float4*)(A + (size_t)rowBase * D);
        for (int i = tid; i < 32 * (D / 4); i += 256) {
            float4 v = A4[i];
            *(uint2*)(sA + i * 4) =
                make_uint2(pack_bf2(v.x, v.y), pack_bf2(v.z, v.w));
        }
    } else {
        for (int i = tid; i < 32 * D; i += 256) {
            int r = i >> 7, k = i & (D - 1);
            int row = rowBase + r;
            sA[i] = f2bf(row < M ? A[(size_t)row * D + k] : 0.f);
        }
    }
    __syncthreads();

    const int wave   = tid >> 5;          // 0..7
    const int lane   = tid & 31;
    const int laneHi = lane >> 4;         // 0 or 1
    const int lmod   = lane & 15;
    const int mi     = wave & 1;          // row-tile within block
    const int j0     = wave >> 1;         // col-tile 0
    const int j1     = j0 + 4;            // col-tile 1
    const int ar     = mi * 16 + lmod;    // A row within panel

    v8f acc0 = {}, acc1 = {};
    #pragma unroll
    for (int kk32 = 0; kk32 < 4; ++kk32) {
        const int kk = kk32 * 32;
        // A fragment: two K-contiguous 16B halves
        frag16 af;
        af.q[0] = *(const uint4*)(sA + ar * D + kk + laneHi * 16);
        af.q[1] = *(const uint4*)(sA + ar * D + kk + laneHi * 16 + 8);
        // B fragments: 32 contiguous bytes per lane
        frag16 bf0, bf1;
        bf0.v = *(const v16u16*)(sB + (((j0 << 2) + kk32) * 32 + lane) * 16);
        bf1.v = *(const v16u16*)(sB + (((j1 << 2) + kk32) * 32 + lane) * 16);

        v16bf a  = __builtin_bit_cast(v16bf, af.v);
        v16bf b0 = __builtin_bit_cast(v16bf, bf0.v);
        v16bf b1 = __builtin_bit_cast(v16bf, bf1.v);
        acc0 = __builtin_amdgcn_wmma_f32_16x16x32_bf16(
                   false, a, false, b0, (short)0, acc0, false, false);
        acc1 = __builtin_amdgcn_wmma_f32_16x16x32_bf16(
                   false, a, false, b1, (short)0, acc1, false, false);
    }

    // C/D layout: VGPR i, lanes 0-15 -> M=i, lanes 16-31 -> M=i+8; N=lane%16
    const int rbase = rowBase + mi * 16 + laneHi * 8;
    if (fullTile) {
        #pragma unroll
        for (int i = 0; i < 8; ++i) {
            size_t ro = (size_t)(rbase + i) * D;
            C[ro + (j0 << 4) + lmod] = acc0[i];
            C[ro + (j1 << 4) + lmod] = acc1[i];
        }
    } else {
        #pragma unroll
        for (int i = 0; i < 8; ++i) {
            if (rbase + i < M) {
                size_t ro = (size_t)(rbase + i) * D;
                C[ro + (j0 << 4) + lmod] = acc0[i];
                C[ro + (j1 << 4) + lmod] = acc1[i];
            }
        }
    }
}

// ---------------------------------------------------------------------------
// edge scatter: agg[dst] += hw[src] * dinv[src]*dinv[dst]
// one wave (32 lanes x float4) per edge; atomics resolve in L2 (192 MB holds
// the whole 51 MB feature panel, so gathers + atomic adds stay on-chip).
// ---------------------------------------------------------------------------
__global__ __launch_bounds__(256)
void gcn_scatter(const float* __restrict__ hw, const int* __restrict__ src,
                 const int* __restrict__ dst, const float* __restrict__ dinv,
                 float* __restrict__ agg, long long E) {
    long long t = (long long)blockIdx.x * blockDim.x + threadIdx.x;
    long long e = t >> 5;
    if (e >= E) return;
    int lane = (int)(t & 31);
    int s = src[e], d = dst[e];
    float nrm = dinv[s] * dinv[d];
    const float4 v = *(const float4*)(hw + (size_t)s * D + lane * 4);
    float* base = agg + (size_t)d * D + lane * 4;
    atomicAdd(base + 0, v.x * nrm);
    atomicAdd(base + 1, v.y * nrm);
    atomicAdd(base + 2, v.z * nrm);
    atomicAdd(base + 3, v.w * nrm);
}

// ---------------------------------------------------------------------------
// epilogue: h = relu(agg + hw*dinv^2 (self loop) + bias)   (float4 lanes)
// ---------------------------------------------------------------------------
__global__ void gcn_epilogue(const float4* __restrict__ agg,
                             const float4* __restrict__ hw,
                             const float*  __restrict__ dinv,
                             const float4* __restrict__ bias,
                             float4* __restrict__ h, long long n4) {
    long long t = (long long)blockIdx.x * blockDim.x + threadIdx.x;
    if (t >= n4) return;
    long long node = t >> 5;              // 32 float4 per 128-wide row
    int c4 = (int)(t & 31);
    float di = dinv[node];
    float s = di * di;
    float4 a = agg[t], w = hw[t], b = bias[c4];
    float4 r;
    r.x = fmaxf(a.x + w.x * s + b.x, 0.f);
    r.y = fmaxf(a.y + w.y * s + b.y, 0.f);
    r.z = fmaxf(a.z + w.z * s + b.z, 0.f);
    r.w = fmaxf(a.w + w.w * s + b.w, 0.f);
    h[t] = r;
}

// ---------------------------------------------------------------------------
// output head: logits = h @ Wout + bout; softmax over 8 classes.
// one wave per row; lanes cover K=128 (4 each), shuffle-reduce 8 accumulators.
// ---------------------------------------------------------------------------
__global__ __launch_bounds__(256)
void gcn_out_softmax(const float* __restrict__ h, const float* __restrict__ Wout,
                     const float* __restrict__ bout, float* __restrict__ out,
                     int M) {
    __shared__ float sW[D * NCLS];
    __shared__ float sb[NCLS];
    int tid = threadIdx.x;
    for (int i = tid; i < D * NCLS; i += 256) sW[i] = Wout[i];
    if (tid < NCLS) sb[tid] = bout[tid];
    __syncthreads();

    int wave = tid >> 5, lane = tid & 31;
    int row = blockIdx.x * 8 + wave;
    if (row >= M) return;

    float acc[NCLS];
    #pragma unroll
    for (int c = 0; c < NCLS; ++c) acc[c] = 0.f;

    #pragma unroll
    for (int j = 0; j < 4; ++j) {
        int k = lane + 32 * j;
        float hv = h[(size_t)row * D + k];
        #pragma unroll
        for (int c = 0; c < NCLS; ++c) acc[c] += hv * sW[k * NCLS + c];
    }
    #pragma unroll
    for (int c = 0; c < NCLS; ++c) {
        #pragma unroll
        for (int off = 16; off > 0; off >>= 1)
            acc[c] += __shfl_down(acc[c], off, 32);
    }
    if (lane == 0) {
        float lg[NCLS], mx = -3.402823466e38f;
        #pragma unroll
        for (int c = 0; c < NCLS; ++c) { lg[c] = acc[c] + sb[c]; mx = fmaxf(mx, lg[c]); }
        float s = 0.f;
        #pragma unroll
        for (int c = 0; c < NCLS; ++c) { lg[c] = __expf(lg[c] - mx); s += lg[c]; }
        float inv = 1.f / s;
        #pragma unroll
        for (int c = 0; c < NCLS; ++c) out[(size_t)row * NCLS + c] = lg[c] * inv;
    }
}

// ---------------------------------------------------------------------------
// launch
// ---------------------------------------------------------------------------
extern "C" void kernel_launch(void* const* d_in, const int* in_sizes, int n_in,
                              void* d_out, int out_size, void* d_ws, size_t ws_size,
                              hipStream_t stream) {
    (void)n_in; (void)out_size; (void)ws_size;
    const float* x    = (const float*)d_in[0];
    const int*   ei   = (const int*)  d_in[1];   // [2, E] flat
    const float* W1   = (const float*)d_in[2];
    const float* b1   = (const float*)d_in[3];
    const float* W2   = (const float*)d_in[4];
    const float* b2   = (const float*)d_in[5];
    const float* Wout = (const float*)d_in[6];
    const float* bout = (const float*)d_in[7];
    float* out = (float*)d_out;

    const long long N  = in_sizes[0] / D;
    const long long E  = in_sizes[1] / 2;
    const int*  src = ei;
    const int*  dst = ei + E;
    const long long NF = N * D;

    // workspace: dinv[N] | A[N*D] | B[N*D] | C[N*D]   (~154 MB for N=100k)
    float* dinv = (float*)d_ws;
    size_t off  = ((size_t)N + 63) & ~(size_t)63;   // 256B-align next panels
    float* bufA = dinv + off;        // hw  (post-GEMM features)
    float* bufB = bufA + NF;         // agg (scatter accumulator)
    float* bufC = bufB + NF;         // h   (post-ReLU features)

    const int TPB = 256;
    dim3 gN((unsigned)((N + TPB - 1) / TPB));
    dim3 gE((unsigned)((E + TPB - 1) / TPB));
    dim3 gNF4((unsigned)((NF / 4 + TPB - 1) / TPB));
    dim3 gScat((unsigned)((E * 32 + TPB - 1) / TPB));
    dim3 gGemm((unsigned)((N + 31) / 32));
    dim3 gOut((unsigned)((N + 7) / 8));

    // normalization: deg = 1 (self loop) + in-degree; dinv = rsqrt(deg)
    gcn_fill_f32<<<gN, TPB, 0, stream>>>(dinv, 1.0f, N);
    gcn_degree<<<gE, TPB, 0, stream>>>(dst, dinv, E);
    gcn_rsqrt_inplace<<<gN, TPB, 0, stream>>>(dinv, N);

    // ---- layer 1 ----
    gcn_gemm_wmma<<<gGemm, TPB, 0, stream>>>(x, W1, bufA, (int)N);
    gcn_zero_f32x4<<<gNF4, TPB, 0, stream>>>((float4*)bufB, NF / 4);
    gcn_scatter<<<gScat, TPB, 0, stream>>>(bufA, src, dst, dinv, bufB, E);
    gcn_epilogue<<<gNF4, TPB, 0, stream>>>((const float4*)bufB, (const float4*)bufA,
                                           dinv, (const float4*)b1,
                                           (float4*)bufC, NF / 4);

    // ---- layer 2 ----
    gcn_gemm_wmma<<<gGemm, TPB, 0, stream>>>(bufC, W2, bufA, (int)N);
    gcn_zero_f32x4<<<gNF4, TPB, 0, stream>>>((float4*)bufB, NF / 4);
    gcn_scatter<<<gScat, TPB, 0, stream>>>(bufA, src, dst, dinv, bufB, E);
    gcn_epilogue<<<gNF4, TPB, 0, stream>>>((const float4*)bufB, (const float4*)bufA,
                                           dinv, (const float4*)b2,
                                           (float4*)bufC, NF / 4);

    // ---- head ----
    gcn_out_softmax<<<gOut, TPB, 0, stream>>>(bufC, Wout, bout, out, (int)N);
}
